// ROSA_QKV_LAYER_21466246545679
// MI455X (gfx1250) — compile-verified, matrix operand order checked by reading, not run on hardware
//
#include <hip/hip_runtime.h>
#include <hip/hip_bf16.h>
#include <stdint.h>

// ---------------- CDNA5 / gfx1250 WMMA types ----------------
typedef _Float16 f16;
typedef __attribute__((ext_vector_type(16))) _Float16 v16h;
typedef __attribute__((ext_vector_type(8)))  float    v8f;

#define B_   4
#define T_   1024
#define D_   512
#define H_   8
#define DH_  64
#define TAU_INV 10.0f

union FragH { v16h v; uint32_t u[8]; f16 h[16]; };

__device__ __forceinline__ v8f wmma_f16(v16h a, v16h b, v8f c) {
  // D = A(16x32 f16) * B(32x16 f16) + C(16x16 f32)
  return __builtin_amdgcn_wmma_f32_16x16x32_f16(
      /*neg_a=*/false, a, /*neg_b=*/false, b,
      /*c_mod=*/(short)0, c, /*reuse_a=*/false, /*reuse_b=*/false);
}

// Load an A (or B-transposed) 16x32 f16 fragment from row-major f16 with
// leading dim `ld`. ISA layout: lane&15 = row; lane>=16 takes K-half +8;
// dwords 0..3 hold K {0,2,4,6}(+1), dwords 4..7 hold K 16+{0,2,4,6}(+1).
__device__ __forceinline__ v16h ld_frag_f16(const f16* tile, int ld, int lane) {
  const int r  = lane & 15;
  const int hf = (lane >> 4) & 1;
  const uint32_t* p = (const uint32_t*)(tile + (size_t)r * ld + hf * 8);
  FragH f;
#pragma unroll
  for (int t = 0; t < 4; ++t) { f.u[t] = p[t]; f.u[4 + t] = p[8 + t]; }
  return f.v;
}

// Same fragment but sourced from f32 memory with on-the-fly f16 convert.
__device__ __forceinline__ v16h ld_fragA_f32(const float* tile, int ld, int lane) {
  const int r  = lane & 15;
  const int hf = (lane >> 4) & 1;
  const float* p = tile + (size_t)r * ld + hf * 8;
  FragH f;
#pragma unroll
  for (int t = 0; t < 4; ++t) {
    float2 a = *(const float2*)(p + 2 * t);
    float2 b = *(const float2*)(p + 16 + 2 * t);
    f.h[2*t]     = (f16)a.x; f.h[2*t + 1] = (f16)a.y;
    f.h[8 + 2*t] = (f16)b.x; f.h[9 + 2*t] = (f16)b.y;
  }
  return f.v;
}

// A-fragment of the (unnormalized) attention probabilities, built directly
// from the scanned scores: p = exp((S[i,j] + j/(i+1))/tau - rowmax), causal.
__device__ __forceinline__ v16h ld_fragA_exp(const float* Srow, int j0, int i,
                                             float mx, float inv_ip1, int lane) {
  const int hf = (lane >> 4) & 1;
  const int jb = j0 + hf * 8;
  FragH f;
#pragma unroll
  for (int t = 0; t < 4; ++t) {
    float2 a = *(const float2*)(Srow + jb + 2 * t);
    float2 b = *(const float2*)(Srow + jb + 16 + 2 * t);
    int j0a = jb + 2 * t, j1a = j0a + 1;
    int j0b = jb + 16 + 2 * t, j1b = j0b + 1;
    f.h[2*t]     = (f16)((j0a <= i) ? __expf((a.x + (float)j0a * inv_ip1) * TAU_INV - mx) : 0.f);
    f.h[2*t + 1] = (f16)((j1a <= i) ? __expf((a.y + (float)j1a * inv_ip1) * TAU_INV - mx) : 0.f);
    f.h[8 + 2*t] = (f16)((j0b <= i) ? __expf((b.x + (float)j0b * inv_ip1) * TAU_INV - mx) : 0.f);
    f.h[9 + 2*t] = (f16)((j1b <= i) ? __expf((b.y + (float)j1b * inv_ip1) * TAU_INV - mx) : 0.f);
  }
  return f.v;
}

// ---------------- K0: weight transposes to f16 (n-major for B fragments) ---
__global__ void k0_prep(const float* wq, const float* wk, const float* wv,
                        const float* wo, f16* wqT, f16* wkT, f16* wvT, f16* xvT) {
  int id = blockIdx.x * 256 + threadIdx.x;
  const int NW = 3 * D_ * D_;
  if (id < NW) {
    int m = id / (D_ * D_);
    int rem = id - m * (D_ * D_);
    int n = rem >> 9, d = rem & 511;
    const float* w = (m == 0) ? wq : (m == 1) ? wk : wv;
    f16* wt = (m == 0) ? wqT : (m == 1) ? wkT : wvT;
    wt[rem] = (f16)w[(size_t)d * D_ + n];           // wt[n][d] = w[d][n]
  } else {
    int id2 = id - NW;
    if (id2 < H_ * 64 * 64) {
      int h = id2 >> 12, rem = id2 & 4095;
      int e = rem >> 6, d = rem & 63;
      // xv = wo.reshape(H, Dh, 64);  xvT[h][e][d] = xv[h][d][e]
      xvT[(size_t)(h * 64 + e) * 64 + d] = (f16)wo[(size_t)(h * 64 + d) * 64 + e];
    }
  }
}

// ---------------- K1: QKV projection + per-head softmax(/tau) --------------
// One wave = 16 rows x one head's 64 cols of one of {q,k,v}.
__global__ void k1_qkv(const float* x, const f16* wqT, const f16* wkT,
                       const f16* wvT, f16* Qsm, f16* Ksm, f16* Vt) {
  const int lane = threadIdx.x & 31;
  const int wave = threadIdx.x >> 5;
  int unit = blockIdx.x * 4 + wave;       // 6144 units
  int tt  = unit & 63;
  int bhm = unit >> 6;
  int bh  = bhm & 31;
  int m   = bhm >> 5;                     // 0=q 1=k 2=v
  int b   = bh >> 3, h = bh & 7;
  const f16* WT = (m == 0) ? wqT : (m == 1) ? wkT : wvT;
  const int t0 = tt * 16;
  const float* xrow = x + (size_t)(b * T_ + t0) * D_;
  const int nc = h * 64;
  v8f c0 = {}, c1 = {}, c2 = {}, c3 = {};
#pragma unroll 4
  for (int k = 0; k < D_; k += 32) {
    v16h a  = ld_fragA_f32(xrow + k, D_, lane);
    v16h b0 = ld_frag_f16(WT + (size_t)(nc +  0) * D_ + k, D_, lane);
    v16h b1 = ld_frag_f16(WT + (size_t)(nc + 16) * D_ + k, D_, lane);
    v16h b2 = ld_frag_f16(WT + (size_t)(nc + 32) * D_ + k, D_, lane);
    v16h b3 = ld_frag_f16(WT + (size_t)(nc + 48) * D_ + k, D_, lane);
    c0 = wmma_f16(a, b0, c0); c1 = wmma_f16(a, b1, c1);
    c2 = wmma_f16(a, b2, c2); c3 = wmma_f16(a, b3, c3);
  }
  // softmax over the 64 head dims of each row (rows live across 16 lanes).
  const int n  = lane & 15;
  const int mo = (lane >> 4) ? 8 : 0;
  f16* qdst = Qsm + (size_t)bh * T_ * 64;
  f16* kdst = Ksm + (size_t)bh * T_ * 64;
  f16* vdst = Vt  + (size_t)bh * 64 * T_;
#pragma unroll
  for (int r = 0; r < 8; ++r) {
    float v0 = c0[r], v1 = c1[r], v2 = c2[r], v3 = c3[r];
    float mx = fmaxf(fmaxf(v0, v1), fmaxf(v2, v3));
    for (int s = 1; s < 16; s <<= 1) mx = fmaxf(mx, __shfl_xor(mx, s, 32));
    float e0 = __expf((v0 - mx) * TAU_INV);
    float e1 = __expf((v1 - mx) * TAU_INV);
    float e2 = __expf((v2 - mx) * TAU_INV);
    float e3 = __expf((v3 - mx) * TAU_INV);
    float sm = e0 + e1 + e2 + e3;
    for (int s = 1; s < 16; s <<= 1) sm += __shfl_xor(sm, s, 32);
    float inv = 1.0f / sm;
    int row = t0 + mo + r;
    if (m == 0) {
      f16* p = qdst + (size_t)row * 64;
      p[ 0 + n] = (f16)(e0 * inv); p[16 + n] = (f16)(e1 * inv);
      p[32 + n] = (f16)(e2 * inv); p[48 + n] = (f16)(e3 * inv);
    } else if (m == 1) {
      f16* p = kdst + (size_t)row * 64;
      p[ 0 + n] = (f16)(e0 * inv); p[16 + n] = (f16)(e1 * inv);
      p[32 + n] = (f16)(e2 * inv); p[48 + n] = (f16)(e3 * inv);
    } else {  // V stored transposed: Vt[bh][d][t]
      vdst[(size_t)( 0 + n) * T_ + row] = (f16)(e0 * inv);
      vdst[(size_t)(16 + n) * T_ + row] = (f16)(e1 * inv);
      vdst[(size_t)(32 + n) * T_ + row] = (f16)(e2 * inv);
      vdst[(size_t)(48 + n) * T_ + row] = (f16)(e3 * inv);
    }
  }
}

// ---------------- K2: causal score GEMM  S = Qsm * Ksm^T -------------------
// Only lower-triangle 64x64 tiles are launched (136 of 256 per bh).
__global__ void k2_scores(const f16* Qsm, const f16* Ksm, float* S) {
  const int lane = threadIdx.x & 31;
  const int wave = threadIdx.x >> 5;
  int blk = blockIdx.x;
  int bh = blk / 136;
  int p  = blk - bh * 136;
  int it = 0;
  while ((it + 1) * (it + 2) / 2 <= p) ++it;
  int jt = p - it * (it + 1) / 2;
  const f16* Q = Qsm + (size_t)bh * T_ * 64;
  const f16* K = Ksm + (size_t)bh * T_ * 64;
  float* Sb = S + (size_t)bh * T_ * T_;
  int ib = it * 64 + wave * 16;
  int jb = jt * 64;
  v8f c0 = {}, c1 = {}, c2 = {}, c3 = {};
#pragma unroll
  for (int k = 0; k < 64; k += 32) {
    v16h a  = ld_frag_f16(Q + (size_t)ib * 64 + k, 64, lane);
    v16h b0 = ld_frag_f16(K + (size_t)(jb +  0) * 64 + k, 64, lane);
    v16h b1 = ld_frag_f16(K + (size_t)(jb + 16) * 64 + k, 64, lane);
    v16h b2 = ld_frag_f16(K + (size_t)(jb + 32) * 64 + k, 64, lane);
    v16h b3 = ld_frag_f16(K + (size_t)(jb + 48) * 64 + k, 64, lane);
    c0 = wmma_f16(a, b0, c0); c1 = wmma_f16(a, b1, c1);
    c2 = wmma_f16(a, b2, c2); c3 = wmma_f16(a, b3, c3);
  }
  const int n  = lane & 15;
  const int mo = (lane >> 4) ? 8 : 0;
  const bool diag = (it == jt);
#pragma unroll
  for (int r = 0; r < 8; ++r) {
    int i = ib + mo + r;
    float* row = Sb + (size_t)i * T_;
    if (!diag) {
      row[jb +  0 + n] = c0[r]; row[jb + 16 + n] = c1[r];
      row[jb + 32 + n] = c2[r]; row[jb + 48 + n] = c3[r];
    } else {  // causal mask: future -> 0 (zeros feed the scan correctly)
      row[jb +  0 + n] = (jb +  0 + n <= i) ? c0[r] : 0.f;
      row[jb + 16 + n] = (jb + 16 + n <= i) ? c1[r] : 0.f;
      row[jb + 32 + n] = (jb + 32 + n <= i) ? c2[r] : 0.f;
      row[jb + 48 + n] = (jb + 48 + n <= i) ? c3[r] : 0.f;
    }
  }
}

// ---------------- K3: per-diagonal scan (rotate+cumsum+cummax collapsed) ---
__global__ void k3_scan(float* S) {
  int idx = blockIdx.x * 256 + threadIdx.x;   // 32768 = 32 bh * 1024 lags
  int bh = idx >> 10;
  int d  = idx & 1023;
  float* Sb = S + (size_t)bh * T_ * T_;
  float cs = 0.f, m = 0.f;                    // leading rotated zeros => m=0 init
  for (int i = d; i < T_; ++i) {
    size_t off = (size_t)i * T_ + (size_t)(i - d);
    float s = Sb[off];
    cs += s;
    m = fmaxf(m, cs * (1.f - s));
    Sb[off] = cs - m;
  }
}

// ---------------- K4: per-row softmax stats of (S + pos)/tau ---------------
__global__ void k4_rowstats(const float* S, float2* rowstat) {
  const int lane = threadIdx.x & 31;
  const int wave = threadIdx.x >> 5;
  int row = blockIdx.x * 4 + wave;            // 32768 rows
  int bh = row >> 10, i = row & 1023;
  const float* Srow = S + (size_t)bh * T_ * T_ + (size_t)i * T_;
  float inv_ip1 = 1.f / (float)(i + 1);
  float mx = -__builtin_inff();
  for (int j = lane; j <= i; j += 32)
    mx = fmaxf(mx, (Srow[j] + (float)j * inv_ip1) * TAU_INV);
  for (int s = 1; s < 32; s <<= 1) mx = fmaxf(mx, __shfl_xor(mx, s, 32));
  float sm = 0.f;
  for (int j = lane; j <= i; j += 32)
    sm += __expf((Srow[j] + (float)j * inv_ip1) * TAU_INV - mx);
  for (int s = 1; s < 32; s <<= 1) sm += __shfl_xor(sm, s, 32);
  if (lane == 0) rowstat[row] = make_float2(mx, 1.f / sm);
}

// ---------------- K5: fused P*V (exp built into A frags) + head out-proj ---
__global__ void k5_attn_out(const float* S, const float2* rowstat, const f16* Vt,
                            const f16* xvT, float* out) {
  __shared__ f16 lds[4][16][72];
  const int lane = threadIdx.x & 31;
  const int wave = threadIdx.x >> 5;
  int blk = blockIdx.x;                       // 512 = 32 bh * 16 row-tiles
  int bh = blk >> 4;
  int it = blk & 15;
  int b = bh >> 3, h = bh & 7;
  int rb = it * 64 + wave * 16;
  const float*  Sb = S  + (size_t)bh * T_ * T_;
  const f16*    Vb = Vt + (size_t)bh * 64 * T_;
  const float2* rs = rowstat + (size_t)bh * T_;
  const int ia = rb + (lane & 15);            // this lane's A row
  const float mxA = rs[ia].x;
  const float inv_ip1 = 1.f / (float)(ia + 1);
  const float* SrowA = Sb + (size_t)ia * T_;
  v8f c0 = {}, c1 = {}, c2 = {}, c3 = {};
  const int cmax = (rb + 15) >> 5;            // causal: skip chunks past diag
  for (int ck = 0; ck <= cmax; ++ck) {
    int j0 = ck * 32;
    v16h a  = ld_fragA_exp(SrowA, j0, ia, mxA, inv_ip1, lane);
    v16h b0 = ld_frag_f16(Vb + (size_t) 0 * T_ + j0, T_, lane);
    v16h b1 = ld_frag_f16(Vb + (size_t)16 * T_ + j0, T_, lane);
    v16h b2 = ld_frag_f16(Vb + (size_t)32 * T_ + j0, T_, lane);
    v16h b3 = ld_frag_f16(Vb + (size_t)48 * T_ + j0, T_, lane);
    c0 = wmma_f16(a, b0, c0); c1 = wmma_f16(a, b1, c1);
    c2 = wmma_f16(a, b2, c2); c3 = wmma_f16(a, b3, c3);
  }
  // normalize rows (1/sum) and stage o-tile to LDS as f16
  const int n  = lane & 15;
  const int mo = (lane >> 4) ? 8 : 0;
#pragma unroll
  for (int r = 0; r < 8; ++r) {
    float inv = rs[rb + mo + r].y;
    lds[wave][mo + r][ 0 + n] = (f16)(c0[r] * inv);
    lds[wave][mo + r][16 + n] = (f16)(c1[r] * inv);
    lds[wave][mo + r][32 + n] = (f16)(c2[r] * inv);
    lds[wave][mo + r][48 + n] = (f16)(c3[r] * inv);
  }
  __syncthreads();
  // out_tile(16x64) = o(16x64) @ xv[h](64x64), B from pre-transposed xvT
  const f16* xb = xvT + (size_t)h * 64 * 64;
  v8f d0 = {}, d1 = {}, d2 = {}, d3 = {};
#pragma unroll
  for (int k = 0; k < 64; k += 32) {
    v16h a  = ld_frag_f16(&lds[wave][0][0] + k, 72, lane);
    v16h e0 = ld_frag_f16(xb + (size_t) 0 * 64 + k, 64, lane);
    v16h e1 = ld_frag_f16(xb + (size_t)16 * 64 + k, 64, lane);
    v16h e2 = ld_frag_f16(xb + (size_t)32 * 64 + k, 64, lane);
    v16h e3 = ld_frag_f16(xb + (size_t)48 * 64 + k, 64, lane);
    d0 = wmma_f16(a, e0, d0); d1 = wmma_f16(a, e1, d1);
    d2 = wmma_f16(a, e2, d2); d3 = wmma_f16(a, e3, d3);
  }
#pragma unroll
  for (int r = 0; r < 8; ++r) {
    int trow = rb + mo + r;
    float* op = out + (size_t)(b * T_ + trow) * D_ + h * 64;
    op[ 0 + n] = d0[r]; op[16 + n] = d1[r];
    op[32 + n] = d2[r]; op[48 + n] = d3[r];
  }
}

// ---------------- host launcher --------------------------------------------
extern "C" void kernel_launch(void* const* d_in, const int* in_sizes, int n_in,
                              void* d_out, int out_size, void* d_ws, size_t ws_size,
                              hipStream_t stream) {
  const float* x  = (const float*)d_in[0];
  const float* wq = (const float*)d_in[1];
  const float* wk = (const float*)d_in[2];
  const float* wv = (const float*)d_in[3];
  const float* wo = (const float*)d_in[4];
  float* out = (float*)d_out;
  uint8_t* ws = (uint8_t*)d_ws;
  // workspace layout
  f16*    wqT     = (f16*)(ws + 0);                    // 512 KiB
  f16*    wkT     = (f16*)(ws + (512ull << 10));       // 512 KiB
  f16*    wvT     = (f16*)(ws + (1024ull << 10));      // 512 KiB
  f16*    xvT     = (f16*)(ws + (1536ull << 10));      // 64 KiB
  f16*    Qsm     = (f16*)(ws + (2ull  << 20));        // 4 MiB
  f16*    Ksm     = (f16*)(ws + (6ull  << 20));        // 4 MiB
  f16*    Vt      = (f16*)(ws + (10ull << 20));        // 4 MiB
  float2* rowstat = (float2*)(ws + (14ull << 20));     // 256 KiB
  float*  S       = (float*)(ws + (16ull << 20));      // 128 MiB (L2-resident)

  k0_prep    <<<(3 * D_ * D_ + H_ * 64 * 64 + 255) / 256, 256, 0, stream>>>(
                 wq, wk, wv, wo, wqT, wkT, wvT, xvT);
  k1_qkv     <<<1536, 128, 0, stream>>>(x, wqT, wkT, wvT, Qsm, Ksm, Vt);
  k2_scores  <<<32 * 136, 128, 0, stream>>>(Qsm, Ksm, S);
  k3_scan    <<<128, 256, 0, stream>>>(S);
  k4_rowstats<<<8192, 128, 0, stream>>>(S, rowstat);
  k5_attn_out<<<512, 128, 0, stream>>>(S, rowstat, Vt, xvT, out);
}
